// TAWRMAC_75600014344285
// MI455X (gfx1250) — matrix operand order repaired and checked by reading, not compiled, not run on hardware
//
#include <hip/hip_runtime.h>
#include <hip/hip_bf16.h>

// ---------------------------------------------------------------- constants
#define N_NODES 100000
#define N_EDGES 500000
#define D   172
#define DP  176      // fp32 row pad (multiple of 16)
#define DPK 192      // bf16 K-padded row (multiple of 32)
#define KNB 20       // neighbors
#define BB  500
#define MSG 688
#define MSGP 704     // multiple of 32
#define QD  344      // Q_DIM
#define QP  352      // multiple of 32
#define KD  516      // K_DIM
#define KP  544      // multiple of 32
#define G3P 528      // 3*D padded to 3*176 (gate-permuted col space)
#define M1  42000    // layer-1 rows: 2000 batch + 40000 neighbors
#define M2  2000

typedef __attribute__((ext_vector_type(16))) __bf16 v16bf;
typedef __attribute__((ext_vector_type(8)))  float  v8f;

union BF16x16 { uint4 u[2]; v16bf v; };

__device__ __forceinline__ unsigned short f2bf(float f) {
  union { float f; unsigned u; } x; x.f = f;
  unsigned r = x.u + 0x7FFFu + ((x.u >> 16) & 1u);   // round-to-nearest-even
  return (unsigned short)(r >> 16);
}
__device__ __forceinline__ float bf2f(unsigned short s) {
  union { unsigned u; float f; } x; x.u = ((unsigned)s) << 16; return x.f;
}
__device__ __forceinline__ uint2 pack4(float a, float b, float c, float d) {
  return make_uint2((unsigned)f2bf(a) | ((unsigned)f2bf(b) << 16),
                    (unsigned)f2bf(c) | ((unsigned)f2bf(d) << 16));
}
__device__ __forceinline__ float sigm(float x) { return 1.f / (1.f + __expf(-x)); }

// Async 16B global -> LDS copy (CDNA5, tracked with ASYNCcnt).
// VDST holds the wave-relative LDS byte offset; generic pointers to __shared__
// carry the LDS offset in their low 32 bits.
__device__ __forceinline__ void async_copy_b128(void* lds_ptr, const void* gptr) {
  unsigned lds_off = (unsigned)(size_t)lds_ptr;
  asm volatile("global_load_async_to_lds_b128 %0, %1, off"
               :: "v"(lds_off), "v"((unsigned long long)gptr)
               : "memory");
}
__device__ __forceinline__ void wait_async0() {
  asm volatile("s_wait_asynccnt 0" ::: "memory");
}

// A operand (16x32 bf16) from LDS row-major tile (contiguous K, matches load_b):
// lanes 0-15 : row=lane,    K = k0..k0+15
// lanes 16-31: row=lane-16, K = k0+16..k0+31
__device__ __forceinline__ v16bf load_a(const unsigned short* lds, int stride, int k0) {
  int lane = threadIdx.x & 31;
  int row  = lane & 15;
  int koff = k0 + ((lane >> 4) ? 16 : 0);
  const unsigned short* p = lds + row * stride + koff;
  BF16x16 t;
  t.u[0] = *(const uint4*)(p);
  t.u[1] = *(const uint4*)(p + 8);
  return t.v;
}

// B operand (32x16 bf16) from pre-packed global weights:
// lanes 0-15 hold n=lane, K=0..15; lanes 16-31 hold n=lane-16, K=16..31.
// Packed dword layout: ((kt*NT+nt)*32 + lane)*8 + j  ->  (K = 2j, 2j+1).
__device__ __forceinline__ v16bf load_b(const unsigned short* pk, int NT, int kt, int nt) {
  int lane = threadIdx.x & 31;
  const uint4* p = (const uint4*)(pk + ((size_t)((kt * NT + nt) * 32 + lane)) * 16);
  BF16x16 t; t.u[0] = p[0]; t.u[1] = p[1];
  return t.v;
}

__device__ __forceinline__ v8f wmma_bf(v16bf a, v16bf b, v8f c) {
  return __builtin_amdgcn_wmma_f32_16x16x32_bf16(false, a, false, b, (short)0, c, false, false);
}

// ---------------------------------------------------------------- pack weights
// src fp32 (rows x cols) row-major -> bf16 B-tile-packed, zero padded.
// gate_perm: permute output cols so gate tiles (r,z,n) of one 16-col group are
// adjacent: packed tile ct = 3*g + slot  ->  src col = slot*172 + g*16 + off
__global__ void __launch_bounds__(256)
pack_b_kernel(const float* __restrict__ src, int rows, int cols,
              unsigned int* __restrict__ dst, int KT, int NT, int gate_perm) {
  size_t total = (size_t)KT * NT * 32 * 8;
  for (size_t i = (size_t)blockIdx.x * blockDim.x + threadIdx.x; i < total;
       i += (size_t)gridDim.x * blockDim.x) {
    int j = i & 7; int l = (int)((i >> 3) & 31); size_t tile = i >> 8;
    int nt = (int)(tile % NT); int kt = (int)(tile / NT);
    int n  = nt * 16 + (l & 15);
    int k  = kt * 32 + ((l >> 4) * 16) + 2 * j;
    int sn; bool nvalid;
    if (gate_perm) {
      int ct = n >> 4, off = n & 15;
      int g = ct / 3, slot = ct % 3;
      sn = slot * D + g * 16 + off;
      nvalid = (g * 16 + off) < D;
    } else { sn = n; nvalid = n < cols; }
    float f0 = (nvalid && k     < rows) ? src[(size_t)k       * cols + sn] : 0.f;
    float f1 = (nvalid && (k+1) < rows) ? src[(size_t)(k + 1) * cols + sn] : 0.f;
    dst[i] = (unsigned)f2bf(f0) | ((unsigned)f2bf(f1) << 16);
  }
}

// ---------------------------------------------------------------- q_time = cos(b)
__global__ void qtime_kernel(const float* __restrict__ time_b, unsigned short* __restrict__ qtime_bf) {
  int c = threadIdx.x;
  if (c < DPK) qtime_bf[c] = (c < D) ? f2bf(cosf(time_b[c])) : (unsigned short)0;
}

// ---------------------------------------------------------------- node/ts lists
__device__ __forceinline__ int nodes_all_at(const int* s, const int* d, const int* ns,
                                            const int* nd, int p) {
  int g = p / BB, r = p % BB;
  return g == 0 ? s[r] : g == 1 ? d[r] : g == 2 ? ns[r] : nd[r];
}
__global__ void __launch_bounds__(256)
build_l1_kernel(const int* s, const int* d, const int* ns, const int* nd,
                const float* et, const int* neigh_idx, const float* neigh_ts,
                int* l1_node, float* l1_ts) {
  int m = blockIdx.x * blockDim.x + threadIdx.x;
  if (m >= M1) return;
  if (m < M2) {
    l1_node[m] = nodes_all_at(s, d, ns, nd, m);
    l1_ts[m]   = et[m % BB];
  } else {
    int j = m - M2, p = j / KNB, k = j % KNB;
    int pn = nodes_all_at(s, d, ns, nd, p);
    l1_node[m] = neigh_idx[(size_t)pn * KNB + k];
    l1_ts[m]   = neigh_ts[(size_t)pn * KNB + k];
  }
}

// ---------------------------------------------------------------- GRU + base
// 16 node rows per block, 8 waves. Gate nonlinearity fused in-register via the
// gate-permuted Wi/Wh packing (6 WMMA accumulators per gate column-tile).
__global__ void __launch_bounds__(256)
gru_kernel(const float* __restrict__ messages, const float* __restrict__ memory,
           const float* __restrict__ node_features,
           const unsigned short* __restrict__ wi_p, const unsigned short* __restrict__ wh_p,
           const float* __restrict__ bi, const float* __restrict__ bh,
           float* __restrict__ base_f32, unsigned short* __restrict__ base_bf) {
  __shared__ unsigned short a_msg[16 * 712];
  __shared__ unsigned short a_mem[16 * 200];
  int node0 = blockIdx.x * 16;
  int tid = threadIdx.x;
  // vectorized staging: float4 loads -> packed bf16 uint2 stores
  for (int i = tid; i < 16 * (MSGP / 4); i += 256) {
    int r = i / (MSGP / 4), c4 = i % (MSGP / 4), c = c4 * 4;
    uint2 out = make_uint2(0u, 0u);
    if (c < MSG && node0 + r < N_NODES) {
      float4 v = *(const float4*)(messages + (size_t)(node0 + r) * MSG + c);
      out = pack4(v.x, v.y, v.z, v.w);
    }
    *(uint2*)(a_msg + r * 712 + c) = out;
  }
  for (int i = tid; i < 16 * (DPK / 4); i += 256) {
    int r = i / (DPK / 4), c4 = i % (DPK / 4), c = c4 * 4;
    uint2 out = make_uint2(0u, 0u);
    if (c < D && node0 + r < N_NODES) {
      float4 v = *(const float4*)(memory + (size_t)(node0 + r) * D + c);
      out = pack4(v.x, v.y, v.z, v.w);
    }
    *(uint2*)(a_mem + r * 200 + c) = out;
  }
  __syncthreads();
  int wave = tid >> 5, lane = tid & 31;
  for (int g = wave; g < 11; g += 8) {
    v8f ar = {}, az = {}, an = {}, hr = {}, hz = {}, hn = {};
    for (int kt = 0; kt < MSGP / 32; ++kt) {
      v16bf a = load_a(a_msg, 712, kt * 32);
      ar = wmma_bf(a, load_b(wi_p, 33, kt, 3 * g    ), ar);
      az = wmma_bf(a, load_b(wi_p, 33, kt, 3 * g + 1), az);
      an = wmma_bf(a, load_b(wi_p, 33, kt, 3 * g + 2), an);
    }
    for (int kt = 0; kt < DPK / 32; ++kt) {
      v16bf a = load_a(a_mem, 200, kt * 32);
      hr = wmma_bf(a, load_b(wh_p, 33, kt, 3 * g    ), hr);
      hz = wmma_bf(a, load_b(wh_p, 33, kt, 3 * g + 1), hz);
      hn = wmma_bf(a, load_b(wh_p, 33, kt, 3 * g + 2), hn);
    }
    int off = lane & 15, col = g * 16 + off, rb = (lane >> 4) * 8;
    for (int r = 0; r < 8; ++r) {
      int node = node0 + rb + r;
      if (node >= N_NODES) continue;
      if (col < D) {
        float rg = sigm((ar[r] + bi[col])         + (hr[r] + bh[col]));
        float z  = sigm((az[r] + bi[col + D])     + (hz[r] + bh[col + D]));
        float n  = tanhf((an[r] + bi[col + 2*D])  + rg * (hn[r] + bh[col + 2*D]));
        float mv = (1.f - z) * n + z * memory[(size_t)node * D + col];
        float bv = mv + node_features[(size_t)node * D + col];
        base_f32[(size_t)node * DP  + col] = bv;
        base_bf [(size_t)node * DPK + col] = f2bf(bv);
      } else {
        if (col < DP) base_f32[(size_t)node * DP + col] = 0.f;
        base_bf[(size_t)node * DPK + col] = 0;
      }
    }
  }
  // zero bf16 pad cols 176..191 (gate tiles only cover 0..175)
  for (int i = tid; i < 16 * 16; i += 256) {
    int r = i / 16, c = DP + (i % 16);
    int node = node0 + r;
    if (node < N_NODES) base_bf[(size_t)node * DPK + c] = 0;
  }
}

// ---------------------------------------------------------------- qh = [src|qtime] @ Wq
__global__ void __launch_bounds__(256)
qk_kernel(int M, int mode, const int* __restrict__ l1_node,
          const unsigned short* __restrict__ base_bf, const unsigned short* __restrict__ src_bf,
          const unsigned short* __restrict__ qtime_bf,
          const unsigned short* __restrict__ wq_p, float* __restrict__ qh) {
  __shared__ unsigned short a_q[16 * 360];
  int row0 = blockIdx.x * 16, tid = threadIdx.x;
  for (int i = tid; i < 16 * (QP / 4); i += 256) {
    int r = i / (QP / 4), c4 = i % (QP / 4), c = c4 * 4;
    int m = row0 + r;
    uint2 v = make_uint2(0u, 0u);
    if (m < M) {
      if (c < D) {
        const unsigned short* srow = (mode == 0) ? base_bf + (size_t)l1_node[m] * DPK
                                                 : src_bf  + (size_t)m * DPK;
        v = *(const uint2*)(srow + c);
      } else if (c < QD) v = *(const uint2*)(qtime_bf + (c - D));
    }
    *(uint2*)(a_q + r * 360 + c) = v;
  }
  __syncthreads();
  int wave = tid >> 5, lane = tid & 31;
  for (int ct = wave; ct < QP / 16; ct += 8) {
    v8f acc = {};
    for (int kt = 0; kt < QP / 32; ++kt)
      acc = wmma_bf(load_a(a_q, 360, kt * 32), load_b(wq_p, 22, kt, ct), acc);
    int col = ct * 16 + (lane & 15), rb = (lane >> 4) * 8;
    for (int r = 0; r < 8; ++r) {
      int m = row0 + rb + r;
      if (m < M) qh[(size_t)m * QP + col] = acc[r];
    }
  }
}

// ---------------------------------------------------------------- attention core
// One m-row per 4-wave block.  nemb rows (bf16, 384B each) are DMA'd
// global->LDS with async-to-LDS ops (ASYNCcnt); efeat/time-encode fill runs in
// VALU; kh via WMMA; scores+softmax in VALU; vh via WMMA reusing the kh buffer.
__global__ void __launch_bounds__(128)
att_kernel(int M, int mode, const int* __restrict__ l1_node, const float* __restrict__ l1_ts,
           const int* __restrict__ neigh_idx, const int* __restrict__ neigh_eidx,
           const float* __restrict__ neigh_ts,
           const unsigned short* __restrict__ base_bf, const unsigned short* __restrict__ nemb_bf,
           const float* __restrict__ edge_features,
           const float* __restrict__ time_w, const float* __restrict__ time_b,
           const unsigned short* __restrict__ wk_p, const unsigned short* __restrict__ wv_p,
           const float* __restrict__ qh, float* __restrict__ att_out) {
  __shared__ unsigned short kin[32 * 552];
  __shared__ unsigned short kv [32 * 352];    // kh, then reused for vh (bf16)
  __shared__ float qrow[QD];
  __shared__ int   nb_s[KNB]; __shared__ int ne_s[KNB]; __shared__ float nt_s[KNB];
  __shared__ float scores[2 * KNB]; __shared__ float attnw[2 * KNB];
  int m = blockIdx.x, tid = threadIdx.x;
  int   node = l1_node[m];
  float ts   = l1_ts[m];
  if (tid < KNB) {
    nb_s[tid] = neigh_idx [(size_t)node * KNB + tid];
    ne_s[tid] = neigh_eidx[(size_t)node * KNB + tid];
    nt_s[tid] = neigh_ts  [(size_t)node * KNB + tid];
  }
  for (int c = tid; c < QD; c += 128) qrow[c] = qh[(size_t)m * QP + c];
  __syncthreads();
  // 1) async DMA: neighbor embedding rows (384B = 24 x 16B) straight into LDS
  for (int idx = tid; idx < KNB * 24; idx += 128) {
    int r = idx / 24, i = idx % 24;
    const unsigned short* srow = (mode == 0)
        ? base_bf + (size_t)nb_s[r] * DPK
        : nemb_bf + (size_t)(M2 + m * KNB + r) * DPK;
    async_copy_b128(kin + r * 552 + i * 8, srow + i * 8);
  }
  wait_async0();   // before VALU overwrites the padded tail (cols 172..191)
  // 2) efeat + time-encode fill (vectorized), zero elsewhere
  for (int i = tid; i < 32 * 138; i += 128) {
    int r = i / 138, c = (i % 138) * 4;
    if (r < KNB && c < D) continue;              // covered by async DMA
    uint2 out = make_uint2(0u, 0u);
    if (r < KNB && c < KD) {
      if (c < 2 * D) {
        float4 v = *(const float4*)(edge_features + (size_t)ne_s[r] * D + (c - D));
        out = pack4(v.x, v.y, v.z, v.w);
      } else {
        int t = c - 2 * D;
        float4 w = *(const float4*)(time_w + t);
        float4 b = *(const float4*)(time_b + t);
        float dt = ts - nt_s[r];
        out = pack4(cosf(dt * w.x + b.x), cosf(dt * w.y + b.y),
                    cosf(dt * w.z + b.z), cosf(dt * w.w + b.w));
      }
    }
    *(uint2*)(kin + r * 552 + c) = out;
  }
  __syncthreads();
  int wave = tid >> 5, lane = tid & 31;
  // kh = k_in @ Wk  (2 row-tiles x 22 col-tiles)
  for (int t = wave; t < 44; t += 4) {
    int rt = t / 22, ct = t % 22;
    if (t + 4 < 44)   // warm the next B-tile stream
      __builtin_prefetch(wk_p + ((size_t)(((t + 4) % 22) * 32 + lane)) * 16, 0, 0);
    v8f acc = {};
    const unsigned short* arow = kin + rt * 16 * 552;
    for (int kt = 0; kt < KP / 32; ++kt)
      acc = wmma_bf(load_a(arow, 552, kt * 32), load_b(wk_p, 22, kt, ct), acc);
    int col = ct * 16 + (lane & 15), rb = (lane >> 4) * 8;
    for (int r = 0; r < 8; ++r) kv[(rt * 16 + rb + r) * 352 + col] = f2bf(acc[r]);
  }
  __syncthreads();
  if (tid < 2 * KNB) {
    int h = tid / KNB, k = tid % KNB;
    float s = 0.f;
    for (int c = 0; c < D; ++c) s += qrow[h * D + c] * bf2f(kv[k * 352 + h * D + c]);
    scores[tid] = s * 0.076249285166302f;  // 1/sqrt(172)
  }
  __syncthreads();
  if (tid < 2) {
    int h = tid;
    float mx = -1e30f;
    for (int k = 0; k < KNB; ++k) mx = fmaxf(mx, scores[h * KNB + k]);
    float sum = 0.f;
    for (int k = 0; k < KNB; ++k) { float e = __expf(scores[h * KNB + k] - mx); attnw[h * KNB + k] = e; sum += e; }
    float inv = 1.f / sum;
    for (int k = 0; k < KNB; ++k) attnw[h * KNB + k] *= inv;
  }
  __syncthreads();
  // vh = k_in @ Wv (reuse kv buffer)
  for (int t = wave; t < 44; t += 4) {
    int rt = t / 22, ct = t % 22;
    v8f acc = {};
    const unsigned short* arow = kin + rt * 16 * 552;
    for (int kt = 0; kt < KP / 32; ++kt)
      acc = wmma_bf(load_a(arow, 552, kt * 32), load_b(wv_p, 22, kt, ct), acc);
    int col = ct * 16 + (lane & 15), rb = (lane >> 4) * 8;
    for (int r = 0; r < 8; ++r) kv[(rt * 16 + rb + r) * 352 + col] = f2bf(acc[r]);
  }
  __syncthreads();
  for (int c = tid; c < QP; c += 128) {
    float o = 0.f;
    if (c < QD) {
      int h = c / D;
      for (int k = 0; k < KNB; ++k) o += attnw[h * KNB + k] * bf2f(kv[k * 352 + c]);
    }
    att_out[(size_t)m * QP + c] = o;
  }
}

// ---------------------------------------------------------------- Wo -> fc1 -> fc2
__global__ void __launch_bounds__(256)
post_kernel(int M, int mode, const int* __restrict__ l1_node,
            const float* __restrict__ att_out, const float* __restrict__ src_f32,
            const unsigned short* __restrict__ wo_p, const unsigned short* __restrict__ fc1_p,
            const unsigned short* __restrict__ fc2_p,
            const float* __restrict__ fc1_b, const float* __restrict__ fc2_b,
            float* __restrict__ out_f32, unsigned short* __restrict__ out_bf) {
  __shared__ unsigned short a1[16 * 360];   // att_out bf16
  __shared__ unsigned short a2[16 * 552];   // [o2 | src]
  __shared__ unsigned short a3[16 * 200];   // relu(fc1)
  int row0 = blockIdx.x * 16, tid = threadIdx.x;
  for (int i = tid; i < 16 * (QP / 4); i += 256) {
    int r = i / (QP / 4), c = (i % (QP / 4)) * 4;
    int m = row0 + r;
    uint2 out = make_uint2(0u, 0u);
    if (m < M) {
      float4 v = *(const float4*)(att_out + (size_t)m * QP + c);
      out = pack4(v.x, v.y, v.z, v.w);
    }
    *(uint2*)(a1 + r * 360 + c) = out;
  }
  // src half of a2 (cols 344..543)
  for (int i = tid; i < 16 * ((KP - QD) / 4); i += 256) {
    int r = i / ((KP - QD) / 4), c = QD + (i % ((KP - QD) / 4)) * 4;
    int m = row0 + r;
    uint2 out = make_uint2(0u, 0u);
    if (m < M && c < KD) {
      const float* srow = (mode == 0) ? src_f32 + (size_t)l1_node[m] * DP
                                      : src_f32 + (size_t)m * DP;
      float4 v = *(const float4*)(srow + (c - QD));
      out = pack4(v.x, v.y, v.z, v.w);
    }
    *(uint2*)(a2 + r * 552 + c) = out;
  }
  __syncthreads();
  int wave = tid >> 5, lane = tid & 31;
  // o2 = a1 @ Wo   -> bf16 directly into a2 cols 0..343
  for (int ct = wave; ct < QP / 16; ct += 8) {
    v8f acc = {};
    for (int kt = 0; kt < QP / 32; ++kt)
      acc = wmma_bf(load_a(a1, 360, kt * 32), load_b(wo_p, 22, kt, ct), acc);
    int col = ct * 16 + (lane & 15), rb = (lane >> 4) * 8;
    if (col < QD)
      for (int r = 0; r < 8; ++r) a2[(rb + r) * 552 + col] = f2bf(acc[r]);
  }
  __syncthreads();
  // h = relu(a2 @ fc1 + b1)
  for (int ct = wave; ct < DPK / 16; ct += 8) {
    v8f acc = {};
    for (int kt = 0; kt < KP / 32; ++kt)
      acc = wmma_bf(load_a(a2, 552, kt * 32), load_b(fc1_p, 12, kt, ct), acc);
    int col = ct * 16 + (lane & 15), rb = (lane >> 4) * 8;
    for (int r = 0; r < 8; ++r) {
      float h = acc[r] + (col < D ? fc1_b[col] : 0.f);
      a3[(rb + r) * 200 + col] = f2bf(fmaxf(h, 0.f));
    }
  }
  __syncthreads();
  // emb = a3 @ fc2 + b2
  for (int ct = wave; ct < DPK / 16; ct += 8) {
    v8f acc = {};
    for (int kt = 0; kt < DPK / 32; ++kt)
      acc = wmma_bf(load_a(a3, 200, kt * 32), load_b(fc2_p, 12, kt, ct), acc);
    int col = ct * 16 + (lane & 15), rb = (lane >> 4) * 8;
    for (int r = 0; r < 8; ++r) {
      int m = row0 + rb + r;
      if (m >= M) continue;
      float e = acc[r] + (col < D ? fc2_b[col] : 0.f);
      if (col < DP) out_f32[(size_t)m * DP + col] = (col < D) ? e : 0.f;
      if (out_bf)  out_bf[(size_t)m * DPK + col] = (col < D) ? f2bf(e) : (unsigned short)0;
    }
  }
}

// ---------------------------------------------------------------- affinity MLP
__global__ void __launch_bounds__(256)
aff_kernel(const float* __restrict__ l2, const float* __restrict__ aff1_w,
           const float* __restrict__ aff1_b, const float* __restrict__ aff2_w,
           const float* __restrict__ aff2_b, float* __restrict__ out) {
  __shared__ float x[2 * D];
  __shared__ float hbuf[D];
  __shared__ float red[256];
  int r = blockIdx.x, tid = threadIdx.x;
  const float* a = (r < BB) ? l2 + (size_t)r * DP            : l2 + (size_t)(2 * BB + (r - BB)) * DP;
  const float* b = (r < BB) ? l2 + (size_t)(BB + r) * DP     : l2 + (size_t)(3 * BB + (r - BB)) * DP;
  for (int c = tid; c < D; c += 256) { x[c] = a[c]; x[D + c] = b[c]; }
  __syncthreads();
  for (int j = tid; j < D; j += 256) {
    float s = aff1_b[j];
    for (int i = 0; i < 2 * D; ++i) s += x[i] * aff1_w[(size_t)i * D + j];
    hbuf[j] = fmaxf(s, 0.f);
  }
  __syncthreads();
  float p = 0.f;
  for (int j = tid; j < D; j += 256) p += hbuf[j] * aff2_w[j];
  red[tid] = p;
  __syncthreads();
  for (int s = 128; s > 0; s >>= 1) { if (tid < s) red[tid] += red[tid + s]; __syncthreads(); }
  if (tid == 0) out[r] = sigm(red[0] + aff2_b[0]);
}

// ---------------------------------------------------------------- launch
extern "C" void kernel_launch(void* const* d_in, const int* in_sizes, int n_in,
                              void* d_out, int out_size, void* d_ws, size_t ws_size,
                              hipStream_t stream) {
  const int*   src_nodes  = (const int*)  d_in[0];
  const int*   dst_nodes  = (const int*)  d_in[1];
  const int*   neg_src    = (const int*)  d_in[2];
  const int*   neg_dst    = (const int*)  d_in[3];
  const float* edge_times = (const float*)d_in[4];
  const float* node_feat  = (const float*)d_in[5];
  const float* edge_feat  = (const float*)d_in[6];
  const float* memory     = (const float*)d_in[7];
  const float* messages   = (const float*)d_in[8];
  const int*   neigh_idx  = (const int*)  d_in[9];
  const int*   neigh_eidx = (const int*)  d_in[10];
  const float* neigh_ts   = (const float*)d_in[11];
  const float* time_w     = (const float*)d_in[12];
  const float* time_b     = (const float*)d_in[13];
  const float* gru_Wi     = (const float*)d_in[14];
  const float* gru_Wh     = (const float*)d_in[15];
  const float* gru_bi     = (const float*)d_in[16];
  const float* gru_bh     = (const float*)d_in[17];
  const float* Wq         = (const float*)d_in[18];
  const float* Wk         = (const float*)d_in[19];
  const float* Wv         = (const float*)d_in[20];
  const float* Wo         = (const float*)d_in[21];
  const float* fc1_w      = (const float*)d_in[22];
  const float* fc1_b      = (const float*)d_in[23];
  const float* fc2_w      = (const float*)d_in[24];
  const float* fc2_b      = (const float*)d_in[25];
  const float* aff1_w     = (const float*)d_in[26];
  const float* aff1_b     = (const float*)d_in[27];
  const float* aff2_w     = (const float*)d_in[28];
  const float* aff2_b     = (const float*)d_in[29];

  char* cur = (char*)d_ws;
  auto alloc = [&](size_t n) { void* p = (void*)cur; cur += (n + 255) & ~(size_t)255; return p; };

  // packed bf16 weights (~4 MB)
  unsigned short* wi_p  = (unsigned short*)alloc((size_t)MSGP * G3P * 2);
  unsigned short* wh_p  = (unsigned short*)alloc((size_t)DPK  * G3P * 2);
  unsigned short* wq_p[2], *wk_p[2], *wv_p[2], *wo_p[2], *f1_p[2], *f2_p[2];
  for (int l = 0; l < 2; ++l) {
    wq_p[l] = (unsigned short*)alloc((size_t)QP  * QP  * 2);
    wk_p[l] = (unsigned short*)alloc((size_t)KP  * QP  * 2);
    wv_p[l] = (unsigned short*)alloc((size_t)KP  * QP  * 2);
    wo_p[l] = (unsigned short*)alloc((size_t)QP  * QP  * 2);
    f1_p[l] = (unsigned short*)alloc((size_t)KP  * DPK * 2);
    f2_p[l] = (unsigned short*)alloc((size_t)DPK * DPK * 2);
  }
  unsigned short* qtime_bf = (unsigned short*)alloc(DPK * 2);
  float*          base_f32 = (float*)alloc((size_t)N_NODES * DP  * 4);  // 70.4 MB
  unsigned short* base_bf  = (unsigned short*)alloc((size_t)N_NODES * DPK * 2); // 38.4 MB
  int*            l1_node  = (int*)alloc((size_t)M1 * 4);
  float*          l1_ts    = (float*)alloc((size_t)M1 * 4);
  float*          qh       = (float*)alloc((size_t)M1 * QP * 4);  // 59.1 MB (reused by L2)
  float*          att      = (float*)alloc((size_t)M1 * QP * 4);  // 59.1 MB (reused by L2)
  float*          l1out_f  = (float*)alloc((size_t)M1 * DP * 4);  // 29.6 MB
  unsigned short* l1out_bf = (unsigned short*)alloc((size_t)M1 * DPK * 2);
  float*          l2out_f  = (float*)alloc((size_t)M2 * DP * 4);
  // total ~ 280 MB of d_ws

  // 1) pack weights
  pack_b_kernel<<<256, 256, 0, stream>>>(gru_Wi, MSG, 3*D, (unsigned*)wi_p, MSGP/32, G3P/16, 1);
  pack_b_kernel<<<256, 256, 0, stream>>>(gru_Wh, D,   3*D, (unsigned*)wh_p, DPK/32,  G3P/16, 1);
  for (int l = 0; l < 2; ++l) {
    pack_b_kernel<<<256, 256, 0, stream>>>(Wq + (size_t)l*QD*QD, QD, QD, (unsigned*)wq_p[l], QP/32, QP/16, 0);
    pack_b_kernel<<<256, 256, 0, stream>>>(Wk + (size_t)l*KD*QD, KD, QD, (unsigned*)wk_p[l], KP/32, QP/16, 0);
    pack_b_kernel<<<256, 256, 0, stream>>>(Wv + (size_t)l*KD*QD, KD, QD, (unsigned*)wv_p[l], KP/32, QP/16, 0);
    pack_b_kernel<<<256, 256, 0, stream>>>(Wo + (size_t)l*QD*QD, QD, QD, (unsigned*)wo_p[l], QP/32, QP/16, 0);
    pack_b_kernel<<<256, 256, 0, stream>>>(fc1_w + (size_t)l*KD*D, KD, D, (unsigned*)f1_p[l], KP/32,  DPK/16, 0);
    pack_b_kernel<<<256, 256, 0, stream>>>(fc2_w + (size_t)l*D*D,  D,  D, (unsigned*)f2_p[l], DPK/32, DPK/16, 0);
  }
  qtime_kernel<<<1, 256, 0, stream>>>(time_b, qtime_bf);

  // 2) GRU memory update -> base
  gru_kernel<<<(N_NODES + 15) / 16, 256, 0, stream>>>(messages, memory, node_feat,
      wi_p, wh_p, gru_bi, gru_bh, base_f32, base_bf);

  // 3) node/ts lists for layer 1
  build_l1_kernel<<<(M1 + 255) / 256, 256, 0, stream>>>(src_nodes, dst_nodes, neg_src, neg_dst,
      edge_times, neigh_idx, neigh_ts, l1_node, l1_ts);

  // 4) layer 1 (li = 0) over 42000 rows
  qk_kernel<<<(M1 + 15) / 16, 256, 0, stream>>>(M1, 0, l1_node, base_bf, (const unsigned short*)0,
      qtime_bf, wq_p[0], qh);
  att_kernel<<<M1, 128, 0, stream>>>(M1, 0, l1_node, l1_ts, neigh_idx, neigh_eidx, neigh_ts,
      base_bf, (const unsigned short*)0, edge_feat, time_w, time_b, wk_p[0], wv_p[0], qh, att);
  post_kernel<<<(M1 + 15) / 16, 256, 0, stream>>>(M1, 0, l1_node, att, base_f32,
      wo_p[0], f1_p[0], f2_p[0], fc1_b, fc2_b, l1out_f, l1out_bf);

  // 5) layer 2 (li = 1) over 2000 rows; src/nemb come from layer-1 output
  qk_kernel<<<(M2 + 15) / 16, 256, 0, stream>>>(M2, 1, l1_node, base_bf, l1out_bf,
      qtime_bf, wq_p[1], qh);
  att_kernel<<<M2, 128, 0, stream>>>(M2, 1, l1_node, l1_ts, neigh_idx, neigh_eidx, neigh_ts,
      base_bf, l1out_bf, edge_feat, time_w, time_b, wk_p[1], wv_p[1], qh, att);
  post_kernel<<<(M2 + 15) / 16, 256, 0, stream>>>(M2, 1, l1_node, att, l1out_f,
      wo_p[1], f1_p[1], f2_p[1], fc1_b + D, fc2_b + D, l2out_f, (unsigned short*)0);

  // 6) affinity scores -> d_out (1000 sigmoid scores)
  aff_kernel<<<2 * BB, 256, 0, stream>>>(l2out_f, aff1_w, aff1_b, aff2_w, aff2_b, (float*)d_out);
}
